// WindowedAttention3D_52974126629354
// MI455X (gfx1250) — matrix operand order repaired and checked by reading, not compiled, not run on hardware
//
#include <hip/hip_runtime.h>

// CDNA5 / gfx1250 windowed 3D attention, bf16 WMMA path.
// Prep kernel: W_qkv/W_proj -> bf16, transposed, in d_ws (L2-resident, 2MB).
// Main kernel: one workgroup per window (grid=1024), 256 threads = 8 wave32;
// weight chunks staged to LDS with global_load_async_to_lds_b128 (ASYNCcnt).

typedef __attribute__((ext_vector_type(16))) __bf16 v16bf;
typedef __attribute__((ext_vector_type(8)))  __bf16 v8bf;
typedef __attribute__((ext_vector_type(8)))  float  v8f;
typedef int v4i_vs __attribute__((vector_size(16)));   // matches builtin's pointee type

#define AS1 __attribute__((address_space(1)))
#define AS3 __attribute__((address_space(3)))

#if defined(__HIP_DEVICE_COMPILE__) && __has_builtin(__builtin_amdgcn_global_load_async_to_lds_b128)
#define HAVE_ASYNC_LDS 1
#else
#define HAVE_ASYNC_LDS 0
#endif

__device__ __forceinline__ void async_copy16(const __bf16* g, __bf16* l) {
#if HAVE_ASYNC_LDS
  __builtin_amdgcn_global_load_async_to_lds_b128((AS1 v4i_vs*)g, (AS3 v4i_vs*)l, 0, 0);
#else
  *(v8bf*)l = *(const v8bf*)g;
#endif
}

__device__ __forceinline__ void async_wait_all() {
#if HAVE_ASYNC_LDS
#if __has_builtin(__builtin_amdgcn_s_wait_asynccnt)
  __builtin_amdgcn_s_wait_asynccnt(0);
#else
  asm volatile("s_wait_asynccnt 0" ::: "memory");
#endif
#endif
}

#define PX 520   // sX / sO row pitch (512 + 8) -> 1040B rows, 16B aligned
#define PW 40    // staged-weight pitch (32 k + 8) -> 80B rows, 16B aligned
#define PQ 72    // sQ/sK/sVT/sP pitch (64 + 8) -> 144B rows, 16B aligned
#define PS 68    // score pitch (floats)

constexpr int SM_X  = 0;
constexpr int SM_O  = SM_X  + 64 * PX;
constexpr int SM_W  = SM_O  + 64 * PX;        // 3 x [64 n][PW] staged qkv weights (transposed)
constexpr int SM_Q  = SM_W  + 3 * 64 * PW;
constexpr int SM_K  = SM_Q  + 64 * PQ;
constexpr int SM_VT = SM_K  + 64 * PQ;
constexpr int SM_P  = SM_VT + 64 * PQ;
constexpr int SM_S  = SM_P  + 64 * PQ;        // float region: 64*PS floats
constexpr int SM_WP = SM_S  + 64 * PS * 2;    // 512 x [PW] staged proj weights (transposed)
constexpr int SM_TOT_ELEMS = SM_WP + 512 * PW;
constexpr size_t SMEM_BYTES = (size_t)SM_TOT_ELEMS * 2;   // ~244 KB (< 320 KB/WGP)

__device__ __forceinline__ v8f zero8() {
  v8f z;
  #pragma unroll
  for (int i = 0; i < 8; ++i) z[i] = 0.0f;
  return z;
}

// A operand (16x32 bf16, storage [m][k], k contiguous).
// ISA: lanes 0-15 M=lane, K={0..7,16..23}; lanes 16-31 M=lane-16, K={8..15,24..31}.
__device__ __forceinline__ v16bf load_A(const __bf16* s, int pitch, int m0, int k0, int lane) {
  const int ll = lane & 15, lh = lane >> 4;
  const __bf16* p = s + (m0 + ll) * pitch + k0 + lh * 8;
  v16bf a;
  ((v8bf*)&a)[0] = *(const v8bf*)(p);
  ((v8bf*)&a)[1] = *(const v8bf*)(p + 16);
  return a;
}

// B operand (32x16 bf16) from TRANSPOSED storage [n][k] (k contiguous).
// ISA: lanes 0-15 N=lane, K={0..7,16..23}; lanes 16-31 N=lane-16, K={8..15,24..31}.
__device__ __forceinline__ v16bf load_BT(const __bf16* s, int pitch, int n0, int k0, int lane) {
  const int ll = lane & 15, lh = lane >> 4;
  const __bf16* p = s + (n0 + ll) * pitch + k0 + lh * 8;
  v16bf b;
  ((v8bf*)&b)[0] = *(const v8bf*)(p);
  ((v8bf*)&b)[1] = *(const v8bf*)(p + 16);
  return b;
}

__device__ __forceinline__ v8f wmma_bf16(v16bf a, v16bf b, v8f c) {
  return __builtin_amdgcn_wmma_f32_16x16x32_bf16(false, a, false, b, (short)0, c, false, false);
}

// ---------------- prep: fp32 [K][N] -> bf16 transposed [N][K] (tiled) ----------------
__global__ __launch_bounds__(256)
void preconv_weights_kernel(const float* __restrict__ Wqkv,
                            const float* __restrict__ Wproj,
                            __bf16* __restrict__ WTqkv,
                            __bf16* __restrict__ WTproj) {
  __shared__ float tile[64][65];
  const int tid = threadIdx.x;
  int blk = blockIdx.x;
  const float* W; __bf16* WT; int N, kt, nt;
  if (blk < 192) {                       // Wqkv: K=512, N=1536 -> 8 x 24 tiles
    W = Wqkv; WT = WTqkv; N = 1536; kt = blk / 24; nt = blk % 24;
  } else {                               // Wproj: 512x512 -> 8 x 8 tiles
    blk -= 192; W = Wproj; WT = WTproj; N = 512; kt = blk / 8; nt = blk % 8;
  }
  const int k0 = kt * 64, n0 = nt * 64;
  for (int j = 0; j < 16; ++j) {
    const int e = j * 256 + tid;
    const int r = e >> 6, c = e & 63;                 // coalesced along n
    tile[r][c] = W[(k0 + r) * N + n0 + c];
  }
  __syncthreads();
  for (int j = 0; j < 16; ++j) {
    const int e = j * 256 + tid;
    const int r = e >> 6, c = e & 63;                 // coalesced along k
    WT[(n0 + r) * 512 + k0 + c] = (__bf16)tile[c][r];
  }
}

// ---------------- main kernel ----------------
template <int PRECONV>
__global__ __launch_bounds__(256, 1)
void win_attn3d_wmma_kernel(const float* __restrict__ x,
                            const float* __restrict__ Wqkv,
                            const float* __restrict__ bqkv,
                            const float* __restrict__ Wproj,
                            const float* __restrict__ bproj,
                            const __bf16* __restrict__ WTqkv,
                            const __bf16* __restrict__ WTproj,
                            float* __restrict__ out) {
  extern __shared__ __align__(16) char smem_raw[];
  __bf16* sX  = (__bf16*)smem_raw + SM_X;
  __bf16* sO  = (__bf16*)smem_raw + SM_O;
  __bf16* sW  = (__bf16*)smem_raw + SM_W;
  __bf16* sQ  = (__bf16*)smem_raw + SM_Q;
  __bf16* sK  = (__bf16*)smem_raw + SM_K;
  __bf16* sVT = (__bf16*)smem_raw + SM_VT;
  __bf16* sP  = (__bf16*)smem_raw + SM_P;
  float*  sS  = (float*)((__bf16*)smem_raw + SM_S);
  __bf16* sWp = (__bf16*)smem_raw + SM_WP;

  const int tid  = threadIdx.x;
  const int wave = tid >> 5;
  const int lane = tid & 31;
  const int ll   = lane & 15;
  const int lh   = lane >> 4;
  const int mI   = wave & 3;           // each wave owns one 16-row M strip

  const int win = blockIdx.x;          // 0..1023
  const int bb  = win >> 9;
  const int w9  = win & 511;
  const int zb  = ((w9 >> 6) & 7) << 2;
  const int yb  = ((w9 >> 3) & 7) << 2;
  const int xb  = (w9 & 7) << 2;

  // ---- Phase 0: window tokens -> LDS bf16 (coalesced) ----
  for (int i = 0; i < 128; ++i) {
    const int e = i * 256 + tid;
    const int m = e >> 9, c = e & 511;
    const int dz = m >> 4, dy = (m >> 2) & 3, dx = m & 3;
    const int g = ((((bb * 32) + zb + dz) * 32 + yb + dy) * 32 + xb + dx) * 512 + c;
    sX[m * PX + c] = (__bf16)x[g];
  }
  __syncthreads();

  const float scale = 0.125f;

  for (int h = 0; h < 8; ++h) {
    // ===== GEMM1: Q,K,V = X[64x512] @ W[512x64] (x3); 48 tiles, 6 per wave =====
    v8f acc[6];
    #pragma unroll
    for (int i = 0; i < 6; ++i) acc[i] = zero8();

    for (int ks = 0; ks < 16; ++ks) {
      const int k0 = ks * 32;
      if constexpr (PRECONV) {
        // async bf16 copy: 3 x [64 n][32 k] = 768 x 16B chunks, 3 per thread
        #pragma unroll
        for (int j = 0; j < 3; ++j) {
          const int cj   = j * 256 + tid;       // 0..767
          const int row  = cj >> 2;             // ms*64 + col
          const int part = cj & 3;
          const int ms   = row >> 6;
          const int col  = row & 63;
          async_copy16(WTqkv + (ms * 512 + h * 64 + col) * 512 + k0 + part * 8,
                       sW + row * PW + part * 8);
        }
        async_wait_all();
      } else {
        for (int j = 0; j < 24; ++j) {
          const int e   = j * 256 + tid;
          const int ms  = e >> 11;
          const int r   = e & 2047;
          const int kr  = r >> 6;
          const int col = r & 63;
          sW[(ms * 64 + col) * PW + kr] =
              (__bf16)Wqkv[(k0 + kr) * 1536 + ms * 512 + h * 64 + col];
        }
      }
      __syncthreads();
      const v16bf a = load_A(sX, PX, mI * 16, k0, lane);   // reused by all 6 tiles
      #pragma unroll
      for (int i = 0; i < 6; ++i) {
        const int t  = wave + 8 * i;
        const int ms = t >> 4;
        const int nI = (t & 15) >> 2;
        const v16bf b = load_BT(sW + ms * 64 * PW, PW, nI * 16, 0, lane);
        acc[i] = wmma_bf16(a, b, acc[i]);
      }
      __syncthreads();
    }
    // epilogue: +bias; Q,K natural [m][n]; V transposed [n][m] (packed b32 stores)
    #pragma unroll
    for (int i = 0; i < 6; ++i) {
      const int t  = wave + 8 * i;
      const int ms = t >> 4;
      const int nI = (t & 15) >> 2;
      const int n  = nI * 16 + ll;
      const float bias = bqkv[ms * 512 + h * 64 + n];
      if (ms == 2) {
        #pragma unroll
        for (int r = 0; r < 8; r += 2) {
          const int m = mI * 16 + r + lh * 8;          // even -> 4B aligned
          union { __bf16 hh[2]; unsigned u; } pk;
          pk.hh[0] = (__bf16)(acc[i][r] + bias);
          pk.hh[1] = (__bf16)(acc[i][r + 1] + bias);
          *(unsigned*)&sVT[n * PQ + m] = pk.u;
        }
      } else {
        #pragma unroll
        for (int r = 0; r < 8; ++r) {
          const int m = mI * 16 + r + lh * 8;
          const float v = acc[i][r] + bias;
          if (ms == 0) sQ[m * PQ + n] = (__bf16)v;
          else         sK[m * PQ + n] = (__bf16)v;
        }
      }
    }
    __syncthreads();

    // ===== GEMM2: S = Q @ K^T =====
    v8f accs[2] = { zero8(), zero8() };
    #pragma unroll
    for (int ks = 0; ks < 2; ++ks) {
      const int k0 = ks * 32;
      const v16bf a = load_A(sQ, PQ, mI * 16, k0, lane);
      #pragma unroll
      for (int j = 0; j < 2; ++j) {
        const int nI = (wave + 8 * j) >> 2;
        const v16bf b = load_BT(sK, PQ, nI * 16, k0, lane);
        accs[j] = wmma_bf16(a, b, accs[j]);
      }
    }
    #pragma unroll
    for (int j = 0; j < 2; ++j) {
      const int n = ((wave + 8 * j) >> 2) * 16 + ll;
      #pragma unroll
      for (int r = 0; r < 8; ++r) {
        const int m = mI * 16 + r + lh * 8;
        float v = accs[j][r] * scale;
        v = fminf(fmaxf(v, -10000.0f), 10000.0f);
        sS[m * PS + n] = v;
      }
    }
    __syncthreads();

    // ===== softmax: 4 lanes per row, quad shfl reductions =====
    {
      const int row = tid >> 2, qd = tid & 3;
      float ev[16];
      float mx = -3.4e38f;
      #pragma unroll
      for (int j = 0; j < 16; ++j) {
        ev[j] = sS[row * PS + qd * 16 + j];
        mx = fmaxf(mx, ev[j]);
      }
      mx = fmaxf(mx, __shfl_xor(mx, 1, 32));
      mx = fmaxf(mx, __shfl_xor(mx, 2, 32));
      float sum = 0.0f;
      #pragma unroll
      for (int j = 0; j < 16; ++j) { ev[j] = __expf(ev[j] - mx); sum += ev[j]; }
      sum += __shfl_xor(sum, 1, 32);
      sum += __shfl_xor(sum, 2, 32);
      const float inv = 1.0f / sum;
      #pragma unroll
      for (int j = 0; j < 16; ++j)
        sP[row * PQ + qd * 16 + j] = (__bf16)(ev[j] * inv);
    }
    __syncthreads();

    // ===== GEMM3: O_h = P @ V =====
    v8f acco[2] = { zero8(), zero8() };
    #pragma unroll
    for (int ks = 0; ks < 2; ++ks) {
      const int k0 = ks * 32;
      const v16bf a = load_A(sP, PQ, mI * 16, k0, lane);
      #pragma unroll
      for (int j = 0; j < 2; ++j) {
        const int nI = (wave + 8 * j) >> 2;
        const v16bf b = load_BT(sVT, PQ, nI * 16, k0, lane);
        acco[j] = wmma_bf16(a, b, acco[j]);
      }
    }
    #pragma unroll
    for (int j = 0; j < 2; ++j) {
      const int n = ((wave + 8 * j) >> 2) * 16 + ll;
      #pragma unroll
      for (int r = 0; r < 8; ++r) {
        const int m = mI * 16 + r + lh * 8;
        sO[m * PX + h * 64 + n] = (__bf16)acco[j][r];
      }
    }
    __syncthreads();
  } // heads

  // ===== GEMM4: out = O[64x512] @ Wproj[512x512] + b; 128 tiles, 16 per wave =====
  v8f accp[16];
  #pragma unroll
  for (int i = 0; i < 16; ++i) accp[i] = zero8();

  for (int ks = 0; ks < 16; ++ks) {
    const int k0 = ks * 32;
    if constexpr (PRECONV) {
      // async bf16 copy: 512 rows x 64B = 2048 x 16B chunks, 8 per thread
      #pragma unroll
      for (int j = 0; j < 8; ++j) {
        const int cj   = j * 256 + tid;
        const int row  = cj >> 2;
        const int part = cj & 3;
        async_copy16(WTproj + row * 512 + k0 + part * 8,
                     sWp + row * PW + part * 8);
      }
      async_wait_all();
    } else {
      for (int j = 0; j < 64; ++j) {
        const int e   = j * 256 + tid;
        const int kr  = e >> 9;
        const int col = e & 511;
        sWp[col * PW + kr] = (__bf16)Wproj[(k0 + kr) * 512 + col];
      }
    }
    __syncthreads();
    const v16bf a = load_A(sO, PX, mI * 16, k0, lane);    // reused 16x
    #pragma unroll
    for (int i = 0; i < 16; ++i) {
      const int nI = (wave >> 2) + 2 * i;
      const v16bf b = load_BT(sWp, PW, nI * 16, 0, lane);
      accp[i] = wmma_bf16(a, b, accp[i]);
    }
    __syncthreads();
  }

  // epilogue: bias + window-reverse scatter (f32)
  #pragma unroll
  for (int i = 0; i < 16; ++i) {
    const int nI = (wave >> 2) + 2 * i;
    const int c  = nI * 16 + ll;
    const float bias = bproj[c];
    #pragma unroll
    for (int r = 0; r < 8; ++r) {
      const int m  = mI * 16 + r + lh * 8;
      const int dz = m >> 4, dy = (m >> 2) & 3, dx = m & 3;
      const int g  = ((((bb * 32) + zb + dz) * 32 + yb + dy) * 32 + xb + dx) * 512 + c;
      out[g] = accp[i][r] + bias;
    }
  }
}

extern "C" void kernel_launch(void* const* d_in, const int* in_sizes, int n_in,
                              void* d_out, int out_size, void* d_ws, size_t ws_size,
                              hipStream_t stream) {
  (void)in_sizes; (void)n_in; (void)out_size;
  const float* x     = (const float*)d_in[0];
  const float* Wqkv  = (const float*)d_in[1];
  const float* bqkv  = (const float*)d_in[2];
  const float* Wproj = (const float*)d_in[3];
  const float* bproj = (const float*)d_in[4];
  float* out = (float*)d_out;

  constexpr size_t WS_NEED = (size_t)(1536 * 512 + 512 * 512) * 2;  // 2 MB bf16

  if (ws_size >= WS_NEED) {
    __bf16* WTqkv  = (__bf16*)d_ws;
    __bf16* WTproj = WTqkv + 1536 * 512;
    preconv_weights_kernel<<<256, 256, 0, stream>>>(Wqkv, Wproj, WTqkv, WTproj);
    (void)hipFuncSetAttribute(reinterpret_cast<const void*>(win_attn3d_wmma_kernel<1>),
                              hipFuncAttributeMaxDynamicSharedMemorySize, (int)SMEM_BYTES);
    win_attn3d_wmma_kernel<1><<<1024, 256, SMEM_BYTES, stream>>>(
        x, Wqkv, bqkv, Wproj, bproj, WTqkv, WTproj, out);
  } else {
    (void)hipFuncSetAttribute(reinterpret_cast<const void*>(win_attn3d_wmma_kernel<0>),
                              hipFuncAttributeMaxDynamicSharedMemorySize, (int)SMEM_BYTES);
    win_attn3d_wmma_kernel<0><<<1024, 256, SMEM_BYTES, stream>>>(
        x, Wqkv, bqkv, Wproj, bproj, nullptr, nullptr, out);
  }
}
